// Model_performer_49838800502799
// MI455X (gfx1250) — matrix operand order repaired and checked by reading, not compiled, not run on hardware
//
#include <hip/hip_runtime.h>
#include <math.h>

// ---------------------------------------------------------------------------
// Performer encoder-decoder forward for MI455X (gfx1250, wave32, WMMA).
// All GEMMs lower to v_wmma_f32_16x16x32_bf16 (bf16 A/B, f32 accumulate).
// Weights pre-transposed to bf16 [N][K]; activation-side B operands are
// produced in K-contiguous layouts (kpT, hvT, kvT) so every fragment load is
// a contiguous vector load with zero bounds checks (K padded to mult. of 32).
// Waves compute 32x64 tiles when M % 32 == 0 (B-fragment reuse), else 16x64.
// ---------------------------------------------------------------------------

#define B_ 16
#define L_ENC 720
#define L_DEC 144
#define PRED 96
#define DM 512
#define NH 8
#define BH 128            /* B_ * NH */
#define DH 64
#define DFF 2048
#define MF 256
#define CIN 7
#define MARK 4
#define NCP_MAX 736       /* L_ENC padded to multiple of 32 */
#define ALPHA_FM 0.35355339059327f   /* 64^-0.25 */

typedef __attribute__((ext_vector_type(16))) __bf16 v16bf;
typedef __attribute__((ext_vector_type(8)))  float  v8f;

// --------------------------- fragment loaders ------------------------------

__device__ __forceinline__ v16bf load16(const __bf16* p) {
  return *reinterpret_cast<const v16bf*>(p);   // 32B, alignment guaranteed by layout
}

__device__ __forceinline__ v16bf load16(const float* p) {
  float4 x0 = *reinterpret_cast<const float4*>(p);
  float4 x1 = *reinterpret_cast<const float4*>(p + 4);
  float4 x2 = *reinterpret_cast<const float4*>(p + 8);
  float4 x3 = *reinterpret_cast<const float4*>(p + 12);
  v16bf r;
  r[0]=(__bf16)x0.x; r[1]=(__bf16)x0.y; r[2]=(__bf16)x0.z; r[3]=(__bf16)x0.w;
  r[4]=(__bf16)x1.x; r[5]=(__bf16)x1.y; r[6]=(__bf16)x1.z; r[7]=(__bf16)x1.w;
  r[8]=(__bf16)x2.x; r[9]=(__bf16)x2.y; r[10]=(__bf16)x2.z; r[11]=(__bf16)x2.w;
  r[12]=(__bf16)x3.x; r[13]=(__bf16)x3.y; r[14]=(__bf16)x3.z; r[15]=(__bf16)x3.w;
  return r;
}

// A fragment: 16-bit A layout, half-wave K split (aoff baked into pointer)
__device__ __forceinline__ v16bf loadA(const float* ap) {
  float4 x0 = *reinterpret_cast<const float4*>(ap);
  float4 x1 = *reinterpret_cast<const float4*>(ap + 4);
  float4 x2 = *reinterpret_cast<const float4*>(ap + 8);
  float4 x3 = *reinterpret_cast<const float4*>(ap + 12);
  v16bf af;
  af[0]=(__bf16)x0.x; af[1]=(__bf16)x0.y; af[2]=(__bf16)x0.z; af[3]=(__bf16)x0.w;
  af[4]=(__bf16)x1.x; af[5]=(__bf16)x1.y; af[6]=(__bf16)x1.z; af[7]=(__bf16)x1.w;
  af[8]=(__bf16)x2.x; af[9]=(__bf16)x2.y; af[10]=(__bf16)x2.z; af[11]=(__bf16)x2.w;
  af[12]=(__bf16)x3.x; af[13]=(__bf16)x3.y; af[14]=(__bf16)x3.z; af[15]=(__bf16)x3.w;
  return af;
}

// ---------------------------------------------------------------------------
// WMMA GEMM: C[b] = alpha * A[b] @ B[b] (+bias)(gelu).
// One wave -> (16*MT) x 64 tile (MT = 1 or 2).
// A: fp32, K-contiguous, row stride sAm.  B: bf16 or fp32, K-contiguous,
// row-n' stride sBn (i.e. B supplied as [N][K]).  M % (16*MT) == 0,
// N % 64 == 0, K % 32 == 0 (no bounds checks anywhere).
// transC: store C(m,n) at C[n*ldct + m] (epilogue must be mode 0/alpha-only).
// mode: 0 none, 1 +bias, 2 +bias then exact GELU
// ---------------------------------------------------------------------------
template <typename TB, int MT>
__global__ void wmma_gemm_t(const float* __restrict__ A, const TB* __restrict__ Bm,
                            const float* __restrict__ bias, float* __restrict__ C,
                            int M, int N, int K,
                            long long sAm, long long sAb,
                            long long sBn, long long sBb,
                            long long sCb, int transC, int ldct,
                            float alpha, int mode)
{
  const int wave = threadIdx.x >> 5;
  const int lane = threadIdx.x & 31;
  const int ln16 = lane & 15;
  const int hi   = lane >> 4;            // 0 | 1
  const int tilesN = N >> 6;
  const int tilesM = M / (16 * MT);
  long long tile = (long long)blockIdx.x * 8 + wave;
  if (tile >= (long long)tilesM * tilesN) return;
  const int tm = (int)(tile / tilesN);
  const int tn = (int)(tile % tilesN);
  const float* Ab = A + (long long)blockIdx.y * sAb;
  const TB*    Bb = Bm + (long long)blockIdx.y * sBb;
  float*       Cb = C + (long long)blockIdx.y * sCb;

  const int m0 = tm * (16 * MT), n0 = tn << 6;
  const int aoff = hi ? 16 : 0;          // A contiguous K order (this code's choice): K 0..15 | 16..31
  const int boff = hi ? 16 : 0;          // B contiguous K order (same as A): K 0..15 | 16..31
  const float* ap0 = Ab + (long long)(m0 + ln16) * sAm + aoff;
  const float* ap1 = (MT == 2) ? (ap0 + 16 * sAm) : ap0;
  const TB* bp0 = Bb + (long long)(n0 + ln16) * sBn + boff;
  const TB* bp1 = bp0 + 16 * sBn;
  const TB* bp2 = bp0 + 32 * sBn;
  const TB* bp3 = bp0 + 48 * sBn;

  v8f acc[MT][4] = {};
  for (int kt = 0; kt < K; kt += 32) {
    v16bf b0 = load16(bp0 + kt);
    v16bf b1 = load16(bp1 + kt);
    v16bf b2 = load16(bp2 + kt);
    v16bf b3 = load16(bp3 + kt);
    v16bf a0 = loadA(ap0 + kt);
    acc[0][0] = __builtin_amdgcn_wmma_f32_16x16x32_bf16(false, a0, false, b0, (short)0, acc[0][0], false, false);
    acc[0][1] = __builtin_amdgcn_wmma_f32_16x16x32_bf16(false, a0, false, b1, (short)0, acc[0][1], false, false);
    acc[0][2] = __builtin_amdgcn_wmma_f32_16x16x32_bf16(false, a0, false, b2, (short)0, acc[0][2], false, false);
    acc[0][3] = __builtin_amdgcn_wmma_f32_16x16x32_bf16(false, a0, false, b3, (short)0, acc[0][3], false, false);
    if (MT == 2) {
      v16bf a1 = loadA(ap1 + kt);
      acc[1][0] = __builtin_amdgcn_wmma_f32_16x16x32_bf16(false, a1, false, b0, (short)0, acc[1][0], false, false);
      acc[1][1] = __builtin_amdgcn_wmma_f32_16x16x32_bf16(false, a1, false, b1, (short)0, acc[1][1], false, false);
      acc[1][2] = __builtin_amdgcn_wmma_f32_16x16x32_bf16(false, a1, false, b2, (short)0, acc[1][2], false, false);
      acc[1][3] = __builtin_amdgcn_wmma_f32_16x16x32_bf16(false, a1, false, b3, (short)0, acc[1][3], false, false);
    }
  }

#pragma unroll
  for (int i = 0; i < MT; ++i) {
    const int rbase = m0 + i * 16 + (hi << 3);
#pragma unroll
    for (int j = 0; j < 4; ++j) {
      const int ncol = n0 + j * 16 + ln16;
      if (!transC) {
        float bv = (mode >= 1) ? bias[ncol] : 0.f;
#pragma unroll
        for (int r = 0; r < 8; ++r) {
          float v = acc[i][j][r] * alpha + bv;
          if (mode == 2) v = 0.5f * v * (1.f + erff(v * 0.70710678118f));
          Cb[(long long)(rbase + r) * N + ncol] = v;
        }
      } else {
        float* cp = Cb + (long long)ncol * ldct + rbase;
        float4 lo, hi4;
        lo.x = acc[i][j][0]*alpha; lo.y = acc[i][j][1]*alpha;
        lo.z = acc[i][j][2]*alpha; lo.w = acc[i][j][3]*alpha;
        hi4.x = acc[i][j][4]*alpha; hi4.y = acc[i][j][5]*alpha;
        hi4.z = acc[i][j][6]*alpha; hi4.w = acc[i][j][7]*alpha;
        *reinterpret_cast<float4*>(cp)     = lo;
        *reinterpret_cast<float4*>(cp + 4) = hi4;
      }
    }
  }
}

// ----------------------------- small kernels -------------------------------

__device__ __forceinline__ unsigned hash_u32(unsigned x) {
  x ^= x >> 16; x *= 0x7feb352du; x ^= x >> 15; x *= 0x846ca68bu; x ^= x >> 16;
  return x;
}

// Deterministic Gaussian stand-in for the host-side FAVOR+ ORF constants.
__global__ void init_proj_bf16(__bf16* p, unsigned seed)
{
  int i = blockIdx.x * blockDim.x + threadIdx.x;
  if (i >= MF * DH) return;
  unsigned h1 = hash_u32((unsigned)i * 2654435761u + seed * 97u + 1u);
  unsigned h2 = hash_u32(h1 + 0x9e3779b9u);
  float u1 = ((float)h1 + 1.0f) * 2.3283064e-10f;
  float u2 = (float)h2 * 2.3283064e-10f;
  p[i] = (__bf16)(sqrtf(-2.f * logf(u1)) * cosf(6.2831853f * u2));
}

// WT[n*K + k] = (bf16) W[k*N + n]
__global__ void transpose_bf16(const float* __restrict__ W, __bf16* __restrict__ WT,
                               int K, int N)
{
  long long idx = (long long)blockIdx.x * blockDim.x + threadIdx.x;
  if (idx >= (long long)K * N) return;
  int k = (int)(idx % K);
  int n = (int)(idx / K);
  WT[idx] = (__bf16)W[(long long)k * N + n];
}

__global__ void stats_kernel(const float* __restrict__ x, float* means, float* stdev)
{
  int b = blockIdx.x / CIN, c = blockIdx.x % CIN;
  __shared__ float sm[128], sv[128];
  float s = 0.f, s2 = 0.f;
  for (int l = threadIdx.x; l < L_ENC; l += 128) {
    float v = x[((long long)b * L_ENC + l) * CIN + c];
    s += v; s2 += v * v;
  }
  sm[threadIdx.x] = s; sv[threadIdx.x] = s2; __syncthreads();
  for (int o = 64; o > 0; o >>= 1) {
    if (threadIdx.x < o) { sm[threadIdx.x] += sm[threadIdx.x + o]; sv[threadIdx.x] += sv[threadIdx.x + o]; }
    __syncthreads();
  }
  if (threadIdx.x == 0) {
    float mu = sm[0] * (1.f / L_ENC);
    float var = sv[0] * (1.f / L_ENC) - mu * mu;
    means[b * CIN + c] = mu;
    stdev[b * CIN + c] = sqrtf(var + 1e-5f);
  }
}

__global__ void norm_x(const float* __restrict__ x, const float* means,
                       const float* stdev, float* out)
{
  int idx = blockIdx.x * blockDim.x + threadIdx.x;
  if (idx >= B_ * L_ENC * CIN) return;
  int c = idx % CIN;
  int b = idx / (L_ENC * CIN);
  out[idx] = (x[idx] - means[b * CIN + c]) / stdev[b * CIN + c];
}

// Token conv (circular, k=3, no bias) + analytic sinusoidal PE + time-mark proj
__global__ void embed_kernel(const float* __restrict__ x, const float* __restrict__ mark,
                             const float* __restrict__ convW, const float* __restrict__ timeW,
                             float* __restrict__ out, int Lx)
{
  long long idx = (long long)blockIdx.x * blockDim.x + threadIdx.x;
  if (idx >= (long long)B_ * Lx * DM) return;
  int d = (int)(idx % DM);
  int l = (int)((idx / DM) % Lx);
  int b = (int)(idx / ((long long)DM * Lx));
  int lm = (l - 1 + Lx) % Lx, lp = (l + 1) % Lx;
  const float* xb = x + (long long)b * Lx * CIN;
  float acc = 0.f;
#pragma unroll
  for (int c = 0; c < CIN; ++c) {
    acc += xb[(long long)lm * CIN + c] * convW[(0 * CIN + c) * DM + d];
    acc += xb[(long long)l  * CIN + c] * convW[(1 * CIN + c) * DM + d];
    acc += xb[(long long)lp * CIN + c] * convW[(2 * CIN + c) * DM + d];
  }
  int i2 = d & ~1;
  float freq = expf(-logf(10000.f) * (float)i2 / (float)DM);
  float ang = (float)l * freq;
  acc += (d & 1) ? cosf(ang) : sinf(ang);
  const float* mb = mark + ((long long)b * Lx + l) * MARK;
#pragma unroll
  for (int t = 0; t < MARK; ++t) acc += mb[t] * timeW[t * DM + d];
  out[idx] = acc;
}

// [b, n, h*dh] -> [b, h, n, dh]
__global__ void head_split(const float* __restrict__ in, float* __restrict__ out, int n)
{
  long long idx = (long long)blockIdx.x * blockDim.x + threadIdx.x;
  if (idx >= (long long)BH * n * DH) return;
  int d = (int)(idx & (DH - 1));
  long long r = idx >> 6;
  int nn = (int)(r % n); r /= n;
  int h = (int)(r & (NH - 1));
  int b = (int)(r >> 3);
  out[idx] = in[((long long)b * n + nn) * DM + h * DH + d];
}

// [b, n, h*dh] -> transposed [b, h, dh, ncp] with zero padding n >= nc
__global__ void head_split_T(const float* __restrict__ in, float* __restrict__ outT,
                             int nc, int ncp)
{
  long long idx = (long long)blockIdx.x * blockDim.x + threadIdx.x;
  if (idx >= (long long)BH * DH * ncp) return;
  int nn = (int)(idx % ncp);
  long long r = idx / ncp;
  int d = (int)(r % DH);
  long long bh = r / DH;
  int b = (int)(bh >> 3), h = (int)(bh & 7);
  outT[idx] = (nn < nc) ? in[((long long)b * nc + nn) * DM + h * DH + d] : 0.f;
}

// [b, h, n, dh] (optionally scaled per (b,h,n)) -> [b, n, h*dh]
__global__ void head_merge(const float* __restrict__ in, const float* __restrict__ scale,
                           float* __restrict__ out, int n)
{
  long long idx = (long long)blockIdx.x * blockDim.x + threadIdx.x;
  if (idx >= (long long)B_ * n * DM) return;
  int dd = (int)(idx % DM);
  long long r = idx / DM;
  int nn = (int)(r % n);
  int b = (int)(r / n);
  int h = dd >> 6, d = dd & (DH - 1);
  long long bh = (long long)b * NH + h;
  float v = in[(bh * n + nn) * DH + d];
  if (scale) v *= scale[bh * n + nn];
  out[idx] = v;
}

// Query rows (xd normal layout): diag = 0.0625*sum(x^2); per-row max of xd
__global__ void fmap_rowstat(const float* __restrict__ xd, const float* __restrict__ xhead,
                             float* diag, float* rowmax)
{
  long long row = blockIdx.x;
  __shared__ float red[256];
  int t = threadIdx.x;
  float dv = 0.f;
  if (t < DH) { float v = xhead[row * DH + t]; dv = v * v; }
  red[t] = dv; __syncthreads();
  for (int o = 128; o > 0; o >>= 1) { if (t < o) red[t] += red[t + o]; __syncthreads(); }
  if (t == 0) diag[row] = red[0] * 0.0625f;   // 0.5 * (64^-0.25)^2
  __syncthreads();
  red[t] = xd[row * MF + t];
  __syncthreads();
  for (int o = 128; o > 0; o >>= 1) { if (t < o) red[t] = fmaxf(red[t], red[t + o]); __syncthreads(); }
  if (t == 0) rowmax[row] = red[0];
}

// phi_q = MF^-0.5 * (exp(xd - diag - rowmax) + 1e-4), in place (normal layout)
__global__ void fmap_finalize_q(float* __restrict__ xd, const float* __restrict__ diag,
                                const float* __restrict__ rowmax, int n)
{
  long long idx = (long long)blockIdx.x * blockDim.x + threadIdx.x;
  if (idx >= (long long)BH * n * MF) return;
  long long row = idx >> 8;
  float v = xd[idx];
  xd[idx] = 0.0625f * (expf(v - diag[row] - rowmax[row]) + 1e-4f);  // 256^-0.5
}

// diag for key rows (from head buffer only)
__global__ void diag_only(const float* __restrict__ xhead, float* __restrict__ diag,
                          long long rows)
{
  long long idx = (long long)blockIdx.x * blockDim.x + threadIdx.x;
  if (idx >= rows) return;
  const float* p = xhead + idx * DH;
  float s = 0.f;
#pragma unroll
  for (int d = 0; d < DH; ++d) s += p[d] * p[d];
  diag[idx] = s * 0.0625f;
}

// per-(b,h) max over transposed xd_k [MF][ncp], skipping pad columns
__global__ void headmax_T(const float* __restrict__ kpT, float* headstab, int nc, int ncp)
{
  int bh = blockIdx.x;
  const float* p = kpT + (long long)bh * MF * ncp;
  __shared__ float red[256];
  float m = -1e30f;
  for (int i = threadIdx.x; i < MF * ncp; i += 256) {
    int nn = i % ncp;
    if (nn < nc) m = fmaxf(m, p[i]);
  }
  red[threadIdx.x] = m; __syncthreads();
  for (int o = 128; o > 0; o >>= 1) {
    if (threadIdx.x < o) red[threadIdx.x] = fmaxf(red[threadIdx.x], red[threadIdx.x + o]);
    __syncthreads();
  }
  if (threadIdx.x == 0) headstab[bh] = red[0];
}

// phi_k on transposed layout; pad columns forced to 0 (feeds padded-K GEMM)
__global__ void fmap_finalize_T(float* __restrict__ kpT, const float* __restrict__ diag,
                                const float* __restrict__ headstab, int nc, int ncp)
{
  long long idx = (long long)blockIdx.x * blockDim.x + threadIdx.x;
  if (idx >= (long long)BH * MF * ncp) return;
  int nn = (int)(idx % ncp);
  long long r = idx / ncp;                 // bh*MF + m
  int bh = (int)(r >> 8);
  float v = kpT[idx];
  kpT[idx] = (nn < nc)
      ? 0.0625f * (expf(v - diag[(long long)bh * nc + nn] - headstab[bh]) + 1e-4f)
      : 0.f;
}

// ksum[bh, m] = sum_n kpT[bh, m, n]  (pad columns are zero)
__global__ void colsum_T(const float* __restrict__ kpT, float* ksum, int ncp)
{
  int idx = blockIdx.x * blockDim.x + threadIdx.x;
  if (idx >= BH * MF) return;
  const float* p = kpT + (long long)idx * ncp;
  float s = 0.f;
  for (int i = 0; i < ncp; ++i) s += p[i];
  ksum[idx] = s;
}

__global__ void den_kernel(const float* __restrict__ qp, const float* __restrict__ ksum,
                           float* dscale, int n)
{
  long long idx = (long long)blockIdx.x * blockDim.x + threadIdx.x;
  if (idx >= (long long)BH * n) return;
  int bh = (int)(idx / n);
  const float* q = qp + idx * MF;
  const float* ks = ksum + (long long)bh * MF;
  float s = 0.f;
  for (int m = 0; m < MF; ++m) s += q[m] * ks[m];
  dscale[idx] = 1.f / s;
}

// Causal prefix scan; one block per (b,h); S (256x64) held in VGPRs.
__global__ void causal_scan(const float* __restrict__ qp, const float* __restrict__ kpT,
                            const float* __restrict__ hvT, float* __restrict__ out,
                            int n, int ncp)
{
  int bh = blockIdx.x;
  const float* q  = qp  + (long long)bh * n * MF;
  const float* kT = kpT + (long long)bh * MF * ncp;
  const float* vT = hvT + (long long)bh * DH * ncp;
  float*       o  = out + (long long)bh * n * DH;
  __shared__ float qn_s[256], kn_s[256], sacc[256], red[256], numred[256];
  int t = threadIdx.x;
  int d = t & (DH - 1), mg = t >> 6;
  float Sreg[64];
#pragma unroll
  for (int i = 0; i < 64; ++i) Sreg[i] = 0.f;
  sacc[t] = 0.f;
  __syncthreads();
  for (int step = 0; step < n; ++step) {
    qn_s[t] = q[(long long)step * MF + t];
    kn_s[t] = kT[(long long)t * ncp + step];
    __syncthreads();
    float vd = vT[(long long)d * ncp + step];
    float num = 0.f;
#pragma unroll
    for (int j = 0; j < 64; ++j) {
      int mm = mg * 64 + j;
      float s = Sreg[j] + kn_s[mm] * vd;
      Sreg[j] = s;
      num += qn_s[mm] * s;
    }
    numred[t] = num;
    float snew = sacc[t] + kn_s[t];
    sacc[t] = snew;
    red[t] = qn_s[t] * snew;
    __syncthreads();
    for (int off = 128; off > 0; off >>= 1) {
      if (t < off) red[t] += red[t + off];
      __syncthreads();
    }
    if (t < DH) {
      float nv = numred[t] + numred[64 + t] + numred[128 + t] + numred[192 + t];
      o[(long long)step * DH + t] = nv / (red[0] + 1e-6f);
    }
    __syncthreads();
  }
}

// y = LN(X + R) * g + b ; one block per row, D = 512
__global__ void layernorm_kernel(const float* __restrict__ X, const float* __restrict__ R,
                                 const float* __restrict__ g, const float* __restrict__ bta,
                                 float* __restrict__ out)
{
  long long row = blockIdx.x;
  const float* xr = X + row * DM;
  const float* rr = R ? (R + row * DM) : nullptr;
  __shared__ float red[256];
  int t = threadIdx.x;
  float v0 = xr[t]       + (rr ? rr[t]       : 0.f);
  float v1 = xr[t + 256] + (rr ? rr[t + 256] : 0.f);
  red[t] = v0 + v1; __syncthreads();
  for (int o = 128; o > 0; o >>= 1) { if (t < o) red[t] += red[t + o]; __syncthreads(); }
  float mu = red[0] * (1.f / DM);
  __syncthreads();
  float c0 = v0 - mu, c1 = v1 - mu;
  red[t] = c0 * c0 + c1 * c1; __syncthreads();
  for (int o = 128; o > 0; o >>= 1) { if (t < o) red[t] += red[t + o]; __syncthreads(); }
  float rstd = rsqrtf(red[0] * (1.f / DM) + 1e-5f);
  out[row * DM + t]       = c0 * rstd * g[t]       + bta[t];
  out[row * DM + t + 256] = c1 * rstd * g[t + 256] + bta[t + 256];
}

// Final 512->7 projection on the last PRED rows + RevIN de-normalization
__global__ void final_proj(const float* __restrict__ dec, const float* __restrict__ W,
                           const float* __restrict__ bias, const float* __restrict__ means,
                           const float* __restrict__ stdev, float* __restrict__ out)
{
  int idx = blockIdx.x * blockDim.x + threadIdx.x;
  if (idx >= B_ * PRED * CIN) return;
  int c = idx % CIN;
  int t = (idx / CIN) % PRED;
  int b = idx / (CIN * PRED);
  const float* dr = dec + ((long long)b * L_DEC + (L_DEC - PRED) + t) * DM;
  float acc = bias[c];
  for (int d = 0; d < DM; ++d) acc += dr[d] * W[d * CIN + c];
  out[idx] = acc * stdev[b * CIN + c] + means[b * CIN + c];
}

// ----------------------------- host orchestration ---------------------------

static inline int cdiv(long long a, int b) { return (int)((a + b - 1) / b); }

template <typename TB>
static void gemm_any(hipStream_t st, const float* A, const TB* B, const float* bias,
                     float* C, int M, int N, int K,
                     long long sAm, long long sAb, long long sBn, long long sBb,
                     long long sCb, int batch, int transC, int ldct, float alpha, int mode)
{
  if (M % 32 == 0) {
    long long tiles = (long long)(M / 32) * (N / 64);
    dim3 grid(cdiv(tiles, 8), batch);
    wmma_gemm_t<TB, 2><<<grid, 256, 0, st>>>(A, B, bias, C, M, N, K,
                                             sAm, sAb, sBn, sBb, sCb, transC, ldct, alpha, mode);
  } else {
    long long tiles = (long long)(M / 16) * (N / 64);
    dim3 grid(cdiv(tiles, 8), batch);
    wmma_gemm_t<TB, 1><<<grid, 256, 0, st>>>(A, B, bias, C, M, N, K,
                                             sAm, sAb, sBn, sBb, sCb, transC, ldct, alpha, mode);
  }
}

// Full Performer attention block: out = Wo(attn(x, ctx)) + bo
static void run_attention(hipStream_t st, const float* x, const float* ctx, int n, int nc,
                          const __bf16* WqT, const __bf16* WkT, const __bf16* WvT,
                          const __bf16* WoT, const float* bo, const __bf16* projbf,
                          bool causal,
                          float* bufA, float* out,
                          float* hq, float* hk, float* hvT,
                          float* qp, float* kpT, float* kvT,
                          float* diag, float* rowmax, float* headstab,
                          float* ksum, float* dscale)
{
  const int Mq = B_ * n, Mc = B_ * nc;
  const int ncp = (nc + 31) & ~31;
  // projections (no bias) + head split (V split transposed + zero-padded)
  gemm_any(st, x,   WqT, (const float*)nullptr, bufA, Mq, DM, DM, DM, 0, DM, 0, 0, 1, 0, 0, 1.f, 0);
  head_split<<<cdiv((long long)BH * n * DH, 256), 256, 0, st>>>(bufA, hq, n);
  gemm_any(st, ctx, WkT, (const float*)nullptr, bufA, Mc, DM, DM, DM, 0, DM, 0, 0, 1, 0, 0, 1.f, 0);
  head_split<<<cdiv((long long)BH * nc * DH, 256), 256, 0, st>>>(bufA, hk, nc);
  gemm_any(st, ctx, WvT, (const float*)nullptr, bufA, Mc, DM, DM, DM, 0, DM, 0, 0, 1, 0, 0, 1.f, 0);
  head_split_T<<<cdiv((long long)BH * DH * ncp, 256), 256, 0, st>>>(bufA, hvT, nc, ncp);
  // FAVOR+ feature maps (proj already [N][K] layout in bf16)
  gemm_any(st, hq, projbf, (const float*)nullptr, qp, n, MF, DH,
           DH, (long long)n * DH, DH, 0, (long long)n * MF, BH, 0, 0, ALPHA_FM, 0);
  fmap_rowstat<<<BH * n, 256, 0, st>>>(qp, hq, diag, rowmax);
  fmap_finalize_q<<<cdiv((long long)BH * n * MF, 256), 256, 0, st>>>(qp, diag, rowmax, n);
  gemm_any(st, hk, projbf, (const float*)nullptr, kpT, nc, MF, DH,
           DH, (long long)nc * DH, DH, 0, (long long)MF * ncp, BH, 1, ncp, ALPHA_FM, 0);
  diag_only<<<cdiv((long long)BH * nc, 256), 256, 0, st>>>(hk, diag, (long long)BH * nc);
  headmax_T<<<BH, 256, 0, st>>>(kpT, headstab, nc, ncp);
  fmap_finalize_T<<<cdiv((long long)BH * MF * ncp, 256), 256, 0, st>>>(kpT, diag, headstab, nc, ncp);
  if (!causal) {
    colsum_T<<<cdiv(BH * MF, 256), 256, 0, st>>>(kpT, ksum, ncp);
    den_kernel<<<cdiv((long long)BH * n, 256), 256, 0, st>>>(qp, ksum, dscale, n);
    // kvT[d, m] = sum_n kpT[m, n] * hvT[d, n]   (K = ncp, pad columns zero)
    gemm_any(st, kpT, hvT, (const float*)nullptr, kvT, MF, DH, ncp,
             ncp, (long long)MF * ncp, ncp, (long long)DH * ncp,
             (long long)MF * DH, BH, 1, MF, 1.f, 0);
    // o[n, d] = qp[n, :] @ kv   (B = kvT [d][m], K-contiguous)
    gemm_any(st, qp, kvT, (const float*)nullptr, hq, n, DH, MF,
             MF, (long long)n * MF, MF, (long long)MF * DH,
             (long long)n * DH, BH, 0, 0, 1.f, 0);
    head_merge<<<cdiv((long long)B_ * n * DM, 256), 256, 0, st>>>(hq, dscale, bufA, n);
  } else {
    causal_scan<<<BH, 256, 0, st>>>(qp, kpT, hvT, hq, n, ncp);
    head_merge<<<cdiv((long long)B_ * n * DM, 256), 256, 0, st>>>(hq, nullptr, bufA, n);
  }
  gemm_any(st, bufA, WoT, bo, out, Mq, DM, DM, DM, 0, DM, 0, 0, 1, 0, 0, 1.f, 1);
}

extern "C" void kernel_launch(void* const* d_in, const int* in_sizes, int n_in,
                              void* d_out, int out_size, void* d_ws, size_t ws_size,
                              hipStream_t stream)
{
  (void)in_sizes; (void)n_in; (void)out_size; (void)ws_size;
  const float* x_enc      = (const float*)d_in[0];
  const float* x_mark_enc = (const float*)d_in[1];
  const float* x_dec      = (const float*)d_in[2];
  const float* x_mark_dec = (const float*)d_in[3];
  const float* emb_conv_enc = (const float*)d_in[4];
  const float* emb_time_enc = (const float*)d_in[5];
  const float* emb_conv_dec = (const float*)d_in[6];
  const float* emb_time_dec = (const float*)d_in[7];
  const float* enc_Wq = (const float*)d_in[8];
  const float* enc_Wk = (const float*)d_in[9];
  const float* enc_Wv = (const float*)d_in[10];
  const float* enc_Wo = (const float*)d_in[11];
  const float* enc_bo = (const float*)d_in[12];
  const float* enc_ff1 = (const float*)d_in[13];
  const float* enc_b1  = (const float*)d_in[14];
  const float* enc_ff2 = (const float*)d_in[15];
  const float* enc_b2  = (const float*)d_in[16];
  const float* enc_ln_g = (const float*)d_in[17];
  const float* enc_ln_b = (const float*)d_in[18];
  const float* enc_fg = (const float*)d_in[19];
  const float* enc_fb = (const float*)d_in[20];
  const float* dec_sWq = (const float*)d_in[21];
  const float* dec_sWk = (const float*)d_in[22];
  const float* dec_sWv = (const float*)d_in[23];
  const float* dec_sWo = (const float*)d_in[24];
  const float* dec_sbo = (const float*)d_in[25];
  const float* dec_cWq = (const float*)d_in[26];
  const float* dec_cWk = (const float*)d_in[27];
  const float* dec_cWv = (const float*)d_in[28];
  const float* dec_cWo = (const float*)d_in[29];
  const float* dec_cbo = (const float*)d_in[30];
  const float* dec_ff1 = (const float*)d_in[31];
  const float* dec_b1  = (const float*)d_in[32];
  const float* dec_ff2 = (const float*)d_in[33];
  const float* dec_b2  = (const float*)d_in[34];
  const float* dec_ln_g = (const float*)d_in[35];
  const float* dec_ln_b = (const float*)d_in[36];
  const float* dec_fg = (const float*)d_in[37];
  const float* dec_fb = (const float*)d_in[38];
  const float* proj_w = (const float*)d_in[39];
  const float* proj_b = (const float*)d_in[40];
  float* out = (float*)d_out;

  // ---- workspace carve-up ----
  float* ws = (float*)d_ws;
  size_t off = 0;
  auto alloc = [&](size_t nflt) { float* p = ws + off; off += (nflt + 63) & ~(size_t)63; return p; };
  auto allocbf = [&](size_t nbf) { return (__bf16*)alloc((nbf + 1) / 2); };
  const size_t S_BLD = (size_t)B_ * L_ENC * DM;              // 5,898,240
  const size_t S_QP  = (size_t)BH * L_ENC * MF;              // 23,592,960 (== B*L*DFF)
  const size_t S_KPT = (size_t)BH * MF * NCP_MAX;            // 24,117,248
  const size_t S_HVT = (size_t)BH * DH * NCP_MAX;            // 6,029,312

  __bf16* projbf[4];
  for (int i = 0; i < 4; ++i) projbf[i] = allocbf(MF * DH);
  __bf16* WqT[2]; __bf16* WkT[2]; __bf16* WvT[2]; __bf16* WoT[2];
  __bf16* ff1T[2]; __bf16* ff2T[2];
  for (int i = 0; i < 2; ++i) {
    WqT[i] = allocbf((size_t)DM * DM); WkT[i] = allocbf((size_t)DM * DM);
    WvT[i] = allocbf((size_t)DM * DM); WoT[i] = allocbf((size_t)DM * DM);
    ff1T[i] = allocbf((size_t)DM * DFF); ff2T[i] = allocbf((size_t)DFF * DM);
  }
  __bf16* sWqT = allocbf((size_t)DM * DM); __bf16* sWkT = allocbf((size_t)DM * DM);
  __bf16* sWvT = allocbf((size_t)DM * DM); __bf16* sWoT = allocbf((size_t)DM * DM);
  __bf16* cWqT = allocbf((size_t)DM * DM); __bf16* cWkT = allocbf((size_t)DM * DM);
  __bf16* cWvT = allocbf((size_t)DM * DM); __bf16* cWoT = allocbf((size_t)DM * DM);
  __bf16* dff1T = allocbf((size_t)DM * DFF); __bf16* dff2T = allocbf((size_t)DFF * DM);

  float* means   = alloc(B_ * CIN);
  float* stdev   = alloc(B_ * CIN);
  float* xnorm   = alloc((size_t)B_ * L_ENC * CIN);
  float* enc     = alloc(S_BLD);
  float* bufA    = alloc(S_BLD);
  float* bufB    = alloc(S_BLD);
  float* bufC    = alloc(S_BLD);
  float* hq      = alloc(S_BLD);
  float* hk      = alloc(S_BLD);
  float* hvT     = alloc(S_HVT);
  float* qp      = alloc(S_QP);    // reused as FFN hidden (disjoint phases)
  float* kpT     = alloc(S_KPT);
  float* kvT     = alloc((size_t)BH * MF * DH);
  float* diag    = alloc((size_t)BH * L_ENC);
  float* rowmax  = alloc((size_t)BH * L_ENC);
  float* headstab = alloc(BH);
  float* ksum    = alloc((size_t)BH * MF);
  float* dscale  = alloc((size_t)BH * L_ENC);
  float* dec     = alloc((size_t)B_ * L_DEC * DM);
  float* dx1     = alloc((size_t)B_ * L_DEC * DM);
  float* dx2     = alloc((size_t)B_ * L_DEC * DM);
  float* ffh     = qp;

  // FAVOR+ matrices (host constants in reference; deterministic stand-in)
  unsigned seeds[4] = {0u, 1u, 10u, 11u};
  for (int i = 0; i < 4; ++i)
    init_proj_bf16<<<cdiv(MF * DH, 256), 256, 0, stream>>>(projbf[i], seeds[i]);

  // Pre-transpose + convert all weights to bf16 [N][K]
  const int TDD = cdiv((long long)DM * DM, 256), TDF = cdiv((long long)DM * DFF, 256);
  for (int i = 0; i < 2; ++i) {
    transpose_bf16<<<TDD, 256, 0, stream>>>(enc_Wq + (size_t)i * DM * DM, WqT[i], DM, DM);
    transpose_bf16<<<TDD, 256, 0, stream>>>(enc_Wk + (size_t)i * DM * DM, WkT[i], DM, DM);
    transpose_bf16<<<TDD, 256, 0, stream>>>(enc_Wv + (size_t)i * DM * DM, WvT[i], DM, DM);
    transpose_bf16<<<TDD, 256, 0, stream>>>(enc_Wo + (size_t)i * DM * DM, WoT[i], DM, DM);
    transpose_bf16<<<TDF, 256, 0, stream>>>(enc_ff1 + (size_t)i * DM * DFF, ff1T[i], DM, DFF);
    transpose_bf16<<<TDF, 256, 0, stream>>>(enc_ff2 + (size_t)i * DFF * DM, ff2T[i], DFF, DM);
  }
  transpose_bf16<<<TDD, 256, 0, stream>>>(dec_sWq, sWqT, DM, DM);
  transpose_bf16<<<TDD, 256, 0, stream>>>(dec_sWk, sWkT, DM, DM);
  transpose_bf16<<<TDD, 256, 0, stream>>>(dec_sWv, sWvT, DM, DM);
  transpose_bf16<<<TDD, 256, 0, stream>>>(dec_sWo, sWoT, DM, DM);
  transpose_bf16<<<TDD, 256, 0, stream>>>(dec_cWq, cWqT, DM, DM);
  transpose_bf16<<<TDD, 256, 0, stream>>>(dec_cWk, cWkT, DM, DM);
  transpose_bf16<<<TDD, 256, 0, stream>>>(dec_cWv, cWvT, DM, DM);
  transpose_bf16<<<TDD, 256, 0, stream>>>(dec_cWo, cWoT, DM, DM);
  transpose_bf16<<<TDF, 256, 0, stream>>>(dec_ff1, dff1T, DM, DFF);
  transpose_bf16<<<TDF, 256, 0, stream>>>(dec_ff2, dff2T, DFF, DM);

  // RevIN stats + normalization + encoder embedding
  stats_kernel<<<B_ * CIN, 128, 0, stream>>>(x_enc, means, stdev);
  norm_x<<<cdiv(B_ * L_ENC * CIN, 256), 256, 0, stream>>>(x_enc, means, stdev, xnorm);
  embed_kernel<<<cdiv((long long)S_BLD, 256), 256, 0, stream>>>(xnorm, x_mark_enc,
                                                                emb_conv_enc, emb_time_enc, enc, L_ENC);

  // ---- encoder layers ----
  for (int i = 0; i < 2; ++i) {
    run_attention(stream, enc, enc, L_ENC, L_ENC, WqT[i], WkT[i], WvT[i], WoT[i],
                  enc_bo + (size_t)i * DM, projbf[i], false,
                  bufA, bufB, hq, hk, hvT, qp, kpT, kvT, diag, rowmax, headstab, ksum, dscale);
    layernorm_kernel<<<B_ * L_ENC, 256, 0, stream>>>(bufB, enc,
        enc_ln_g + (size_t)(i * 2 + 0) * DM, enc_ln_b + (size_t)(i * 2 + 0) * DM, bufC);
    gemm_any(stream, bufC, ff1T[i], enc_b1 + (size_t)i * DFF, ffh,
             B_ * L_ENC, DFF, DM, DM, 0, DM, 0, 0, 1, 0, 0, 1.f, 2);
    gemm_any(stream, ffh, ff2T[i], enc_b2 + (size_t)i * DM, bufB,
             B_ * L_ENC, DM, DFF, DFF, 0, DFF, 0, 0, 1, 0, 0, 1.f, 1);
    layernorm_kernel<<<B_ * L_ENC, 256, 0, stream>>>(bufB, bufC,
        enc_ln_g + (size_t)(i * 2 + 1) * DM, enc_ln_b + (size_t)(i * 2 + 1) * DM, enc);
  }
  layernorm_kernel<<<B_ * L_ENC, 256, 0, stream>>>(enc, nullptr, enc_fg, enc_fb, enc);

  // ---- decoder ----
  embed_kernel<<<cdiv((long long)B_ * L_DEC * DM, 256), 256, 0, stream>>>(
      x_dec, x_mark_dec, emb_conv_dec, emb_time_dec, dec, L_DEC);
  run_attention(stream, dec, dec, L_DEC, L_DEC, sWqT, sWkT, sWvT, sWoT, dec_sbo,
                projbf[2], true, bufA, bufB, hq, hk, hvT, qp, kpT, kvT,
                diag, rowmax, headstab, ksum, dscale);
  layernorm_kernel<<<B_ * L_DEC, 256, 0, stream>>>(bufB, dec, dec_ln_g + 0 * DM, dec_ln_b + 0 * DM, dx1);
  run_attention(stream, dx1, enc, L_DEC, L_ENC, cWqT, cWkT, cWvT, cWoT, dec_cbo,
                projbf[3], false, bufA, bufB, hq, hk, hvT, qp, kpT, kvT,
                diag, rowmax, headstab, ksum, dscale);
  layernorm_kernel<<<B_ * L_DEC, 256, 0, stream>>>(bufB, dx1, dec_ln_g + 1 * DM, dec_ln_b + 1 * DM, dx2);
  gemm_any(stream, dx2, dff1T, dec_b1, ffh, B_ * L_DEC, DFF, DM, DM, 0, DM, 0, 0, 1, 0, 0, 1.f, 2);
  gemm_any(stream, ffh, dff2T, dec_b2, bufB, B_ * L_DEC, DM, DFF, DFF, 0, DFF, 0, 0, 1, 0, 0, 1.f, 1);
  layernorm_kernel<<<B_ * L_DEC, 256, 0, stream>>>(bufB, dx2, dec_ln_g + 2 * DM, dec_ln_b + 2 * DM, dec);
  layernorm_kernel<<<B_ * L_DEC, 256, 0, stream>>>(dec, nullptr, dec_fg, dec_fb, dec);

  final_proj<<<cdiv(B_ * PRED * CIN, 256), 256, 0, stream>>>(dec, proj_w, proj_b, means, stdev, out);
}
